// HeadAttetion_80977313399765
// MI455X (gfx1250) — matrix operand order repaired and checked by reading, not compiled, not run on hardware
//
#include <hip/hip_runtime.h>

typedef __attribute__((ext_vector_type(16))) __bf16 v16bf;
typedef __attribute__((ext_vector_type(8)))  __bf16 v8bf;
typedef __attribute__((ext_vector_type(8)))  float  v8f;
typedef int gv4i __attribute__((vector_size(16)));

#define AS1 __attribute__((address_space(1)))
#define AS3 __attribute__((address_space(3)))

#define S_DIM 4096
#define E_DIM 1024

#define TM 128
#define TN 64
#define TK 32
#define LDSK 40   // TK + 8 pad (bf16 elems): 80B row stride -> conflict-free frag reads

#if __has_builtin(__builtin_amdgcn_global_load_async_to_lds_b128)
#define HAVE_ASYNC_BUILTIN 1
#else
#define HAVE_ASYNC_BUILTIN 0
#endif

__device__ __forceinline__ void async_copy_b128(const unsigned short* gsrc,
                                                unsigned short* ldst) {
#if HAVE_ASYNC_BUILTIN
  AS1 gv4i* gp = (AS1 gv4i*)(AS1 void*)(void*)const_cast<unsigned short*>(gsrc);
  AS3 gv4i* lp = (AS3 gv4i*)(AS3 void*)(void*)ldst;
  __builtin_amdgcn_global_load_async_to_lds_b128(gp, lp, 0, 0);
#else
  unsigned lds_off = (unsigned)(size_t)(AS3 unsigned short*)ldst;
  unsigned long long ga = (unsigned long long)gsrc;
  asm volatile("global_load_async_to_lds_b128 %0, %1, off"
               :: "v"(lds_off), "v"(ga) : "memory");
#endif
}

__device__ __forceinline__ void wait_async0() {
#if __has_builtin(__builtin_amdgcn_s_wait_asynccnt)
  __builtin_amdgcn_s_wait_asynccnt(0);
#else
  asm volatile("s_wait_asynccnt 0x0" ::: "memory");
#endif
}

__device__ __forceinline__ unsigned short f32_to_bf16_rne(float f) {
  union { float f; unsigned int u; } c; c.f = f;
  unsigned int u = c.u;
  unsigned int r = u + 0x7FFFu + ((u >> 16) & 1u);  // round-to-nearest-even
  return (unsigned short)(r >> 16);
}

__global__ void cvt_f32_to_bf16(const float* __restrict__ in,
                                unsigned short* __restrict__ out, int n) {
  int i = blockIdx.x * blockDim.x + threadIdx.x;
  int stride = gridDim.x * blockDim.x;
  for (; i < n; i += stride) out[i] = f32_to_bf16_rne(in[i]);
}

// C[M,N] = scale * (A @ Bt^T) + bias ; A:[M,K] bf16 row-major, Bt:[N,K] bf16 row-major.
// Writes bf16 (optionally transposed as [N,M]) and/or f32.
__global__ void __launch_bounds__(256)
gemm_bf16_wmma(const unsigned short* __restrict__ A,
               const unsigned short* __restrict__ Bt,
               const float* __restrict__ bias,
               float scale,
               unsigned short* __restrict__ Cbf,
               float* __restrict__ Cf,
               int M, int N, int K, int transC)
{
  __shared__ __align__(16) unsigned short As[2][TM * LDSK];
  __shared__ __align__(16) unsigned short Bs[2][TN * LDSK];

  const int tid    = threadIdx.x;
  const int lane   = tid & 31;
  const int wave   = tid >> 5;
  const int lane16 = lane & 15;
  const int half   = lane >> 4;

  const int blockM   = blockIdx.y * TM;
  const int blockN   = blockIdx.x * TN;
  const int waveMOff = (wave & 3) * 32;   // 4 waves along M
  const int waveNOff = (wave >> 2) * 32;  // 2 waves along N

  v8f acc[2][2];
  #pragma unroll
  for (int i = 0; i < 2; ++i)
    #pragma unroll
    for (int j = 0; j < 2; ++j)
      acc[i][j] = (v8f){0.f, 0.f, 0.f, 0.f, 0.f, 0.f, 0.f, 0.f};

  const int nK = K / TK;

  // each thread moves 2x16B of A-tile and 1x16B of B-tile, async global->LDS
  const int arow = tid >> 2;          // 0..63
  const int acol = (tid & 3) * 8;     // 0,8,16,24 (bf16 elems)

  auto loadTiles = [&](int kt, int buf) {
    const unsigned short* ag = A + (size_t)blockM * K + (size_t)kt * TK;
    const unsigned short* bg = Bt + (size_t)blockN * K + (size_t)kt * TK;
    #pragma unroll
    for (int i = 0; i < 2; ++i) {
      int row = arow + i * 64;
      async_copy_b128(ag + (size_t)row * K + acol,
                      &As[buf][row * LDSK + acol]);
    }
    async_copy_b128(bg + (size_t)arow * K + acol,
                    &Bs[buf][arow * LDSK + acol]);
    if (kt + 2 < nK) {  // warm the caches ahead of the async engine
      __builtin_prefetch(ag + 2 * TK + (size_t)arow * K + acol, 0, 1);
      __builtin_prefetch(bg + 2 * TK + (size_t)arow * K + acol, 0, 1);
    }
  };

  // 16-bit A-matrix 16x32: lanes0-15 get K {h*8..h*8+7, 16+h*8..+7} with h=lane>>4
  auto loadAFrag = [&](int buf, int m0) -> v16bf {
    const unsigned short* p = &As[buf][(m0 + lane16) * LDSK + half * 8];
    v8bf lo = *(const v8bf*)(p);
    v8bf hi = *(const v8bf*)(p + 16);
    v16bf r;
    #pragma unroll
    for (int i = 0; i < 8; ++i) { r[i] = lo[i]; r[i + 8] = hi[i]; }
    return r;
  };
  // 16-bit B-matrix 32x16: lane -> column n=lane16, K {h*8..h*8+7, 16+h*8..+7} with h=lane>>4
  auto loadBFrag = [&](int buf, int n0) -> v16bf {
    const unsigned short* p = &Bs[buf][(n0 + lane16) * LDSK + half * 8];
    v8bf lo = *(const v8bf*)(p);
    v8bf hi = *(const v8bf*)(p + 16);
    v16bf r;
    #pragma unroll
    for (int i = 0; i < 8; ++i) { r[i] = lo[i]; r[i + 8] = hi[i]; }
    return r;
  };

  loadTiles(0, 0);
  wait_async0();
  __syncthreads();

  for (int kt = 0; kt < nK; ++kt) {
    const int cur = kt & 1;
    if (kt + 1 < nK) loadTiles(kt + 1, cur ^ 1);  // async into other buffer

    v16bf a0 = loadAFrag(cur, waveMOff + 0);
    v16bf a1 = loadAFrag(cur, waveMOff + 16);
    v16bf b0 = loadBFrag(cur, waveNOff + 0);
    v16bf b1 = loadBFrag(cur, waveNOff + 16);

    acc[0][0] = __builtin_amdgcn_wmma_f32_16x16x32_bf16(false, a0, false, b0,
                                                        (short)0, acc[0][0], false, false);
    acc[0][1] = __builtin_amdgcn_wmma_f32_16x16x32_bf16(false, a0, false, b1,
                                                        (short)0, acc[0][1], false, false);
    acc[1][0] = __builtin_amdgcn_wmma_f32_16x16x32_bf16(false, a1, false, b0,
                                                        (short)0, acc[1][0], false, false);
    acc[1][1] = __builtin_amdgcn_wmma_f32_16x16x32_bf16(false, a1, false, b1,
                                                        (short)0, acc[1][1], false, false);

    wait_async0();       // this iteration's async copies (to the other buffer) done
    __syncthreads();     // ..across all waves before anyone reads that buffer
  }

  // C/D layout: VGPR r, lane L -> M = r + 8*(L>=16), N = L%16
  #pragma unroll
  for (int fm = 0; fm < 2; ++fm) {
    #pragma unroll
    for (int fn = 0; fn < 2; ++fn) {
      const int gm0 = blockM + waveMOff + fm * 16 + half * 8;
      const int gn  = blockN + waveNOff + fn * 16 + lane16;
      const float b = bias ? bias[gn] : 0.0f;
      #pragma unroll
      for (int r = 0; r < 8; ++r) {
        const float v = acc[fm][fn][r] * scale + b;
        const int gm = gm0 + r;
        if (transC) {
          Cbf[(size_t)gn * M + gm] = f32_to_bf16_rne(v);
        } else {
          if (Cbf) Cbf[(size_t)gm * N + gn] = f32_to_bf16_rne(v);
          if (Cf)  Cf [(size_t)gm * N + gn] = v;
        }
      }
    }
  }
}

__global__ void __launch_bounds__(256)
softmax_rows(float* __restrict__ out, int cols) {
  __shared__ float red[256];
  const int tid = threadIdx.x;
  float* p = out + (size_t)blockIdx.x * cols;

  float m = -3.402823466e38f;
  for (int i = tid; i < cols; i += 256) m = fmaxf(m, p[i]);
  red[tid] = m; __syncthreads();
  for (int s = 128; s > 0; s >>= 1) {
    if (tid < s) red[tid] = fmaxf(red[tid], red[tid + s]);
    __syncthreads();
  }
  m = red[0]; __syncthreads();

  float sum = 0.f;
  for (int i = tid; i < cols; i += 256) {
    float e = __expf(p[i] - m);
    p[i] = e;
    sum += e;
  }
  red[tid] = sum; __syncthreads();
  for (int s = 128; s > 0; s >>= 1) {
    if (tid < s) red[tid] += red[tid + s];
    __syncthreads();
  }
  const float inv = 1.0f / red[0];
  for (int i = tid; i < cols; i += 256) p[i] *= inv;
}

extern "C" void kernel_launch(void* const* d_in, const int* in_sizes, int n_in,
                              void* d_out, int out_size, void* d_ws, size_t ws_size,
                              hipStream_t stream) {
  (void)in_sizes; (void)n_in; (void)out_size; (void)ws_size;
  const float* x  = (const float*)d_in[0];
  const float* Wq = (const float*)d_in[1];
  const float* bq = (const float*)d_in[2];
  const float* Wk = (const float*)d_in[3];
  const float* bk = (const float*)d_in[4];
  const float* Wv = (const float*)d_in[5];
  const float* bv = (const float*)d_in[6];
  float* out = (float*)d_out;

  const int S = S_DIM, E = E_DIM;

  char* ws = (char*)d_ws;
  size_t off = 0;
  auto carve = [&](size_t bytes) -> void* {
    void* p = ws + off;
    off += (bytes + 255) & ~(size_t)255;
    return p;
  };
  unsigned short* x_bf  = (unsigned short*)carve((size_t)S * E * 2);
  unsigned short* Wq_bf = (unsigned short*)carve((size_t)E * E * 2);
  unsigned short* Wk_bf = (unsigned short*)carve((size_t)E * E * 2);
  unsigned short* Wv_bf = (unsigned short*)carve((size_t)E * E * 2);
  unsigned short* q_bf  = (unsigned short*)carve((size_t)S * E * 2);
  unsigned short* k_bf  = (unsigned short*)carve((size_t)S * E * 2);
  unsigned short* vT_bf = (unsigned short*)carve((size_t)E * S * 2);  // [E,S]
  unsigned short* qk_bf = (unsigned short*)carve((size_t)S * S * 2);  // [S,S]

  // fp32 -> bf16 conversions
  {
    int n = S * E;
    cvt_f32_to_bf16<<<(n + 1023) / 1024, 256, 0, stream>>>(x, x_bf, n);
    n = E * E;
    cvt_f32_to_bf16<<<(n + 1023) / 1024, 256, 0, stream>>>(Wq, Wq_bf, n);
    cvt_f32_to_bf16<<<(n + 1023) / 1024, 256, 0, stream>>>(Wk, Wk_bf, n);
    cvt_f32_to_bf16<<<(n + 1023) / 1024, 256, 0, stream>>>(Wv, Wv_bf, n);
  }

  const dim3 blk(256);

  // q = x @ Wq^T + bq ; k likewise ; v stored transposed as vT[E,S]
  dim3 gQ(E / TN, S / TM);
  gemm_bf16_wmma<<<gQ, blk, 0, stream>>>(x_bf, Wq_bf, bq, 1.0f, q_bf,  nullptr, S, E, E, 0);
  gemm_bf16_wmma<<<gQ, blk, 0, stream>>>(x_bf, Wk_bf, bk, 1.0f, k_bf,  nullptr, S, E, E, 0);
  gemm_bf16_wmma<<<gQ, blk, 0, stream>>>(x_bf, Wv_bf, bv, 1.0f, vT_bf, nullptr, S, E, E, 1);

  // qk = (q @ k^T) / sqrt(S) ; k_bf is already the [N,K] form for B = km^T
  dim3 gQK(S / TN, S / TM);
  gemm_bf16_wmma<<<gQK, blk, 0, stream>>>(q_bf, k_bf, nullptr, 1.0f / 64.0f,
                                          qk_bf, nullptr, S, S, E, 0);

  // pv = qk @ vm ; B = vm[K,N] -> Bt = vT[N,K]; write f32 straight to d_out
  dim3 gPV(E / TN, S / TM);
  gemm_bf16_wmma<<<gPV, blk, 0, stream>>>(qk_bf, vT_bf, nullptr, 1.0f,
                                          nullptr, out, S, E, S, 0);

  // softmax over feature axis, in place on d_out
  softmax_rows<<<S, 256, 0, stream>>>(out, E);
}
